// VoxelResBackBone8x_29274497089671
// MI455X (gfx1250) — matrix operand order repaired and checked by reading, not compiled, not run on hardware
//
#include <hip/hip_runtime.h>

// ---------------------------------------------------------------------------
// Types for CDNA5 WMMA (wave32): D(f32 16x16) = A(bf16 16x32) * B(bf16 32x16) + C
// ---------------------------------------------------------------------------
typedef __attribute__((ext_vector_type(16))) __bf16 v16bf;
typedef __attribute__((ext_vector_type(8)))  float  v8f;

__device__ __forceinline__ unsigned short f2bf(float f) {
  unsigned u = __float_as_uint(f);
  u += 0x7FFFu + ((u >> 16) & 1u);          // round-to-nearest-even
  return (unsigned short)(u >> 16);
}
__device__ __forceinline__ float bf2f(unsigned short h) {
  return __uint_as_float(((unsigned)h) << 16);
}

union Frag {                                 // 32 bytes: 16 bf16 per lane
  struct { uint4 lo, hi; } q;
  v16bf v;
};

// ---------------------------------------------------------------------------
// Utility kernels
// ---------------------------------------------------------------------------
__global__ void zero_k(unsigned int* p, long n) {
  long i = (long)blockIdx.x * blockDim.x + threadIdx.x;
  long stride = (long)gridDim.x * blockDim.x;
  for (; i < n; i += stride) p[i] = 0u;
}

__global__ void scatter_k(const float* __restrict__ feat,
                          const int* __restrict__ coords,
                          unsigned short* __restrict__ in0,   // (41,160,160,8) bf16
                          float* __restrict__ mask0,          // (41,160,160) f32
                          int n) {
  int i = blockIdx.x * blockDim.x + threadIdx.x;
  if (i >= n) return;
  int z = coords[i * 4 + 1], y = coords[i * 4 + 2], x = coords[i * 4 + 3];
  long v = ((long)z * 160 + y) * 160 + x;
  mask0[v] = 1.0f;
  #pragma unroll
  for (int c = 0; c < 4; c++) in0[v * 8 + c] = f2bf(feat[i * 4 + c]);
}

__global__ void mask_down_k(const float* __restrict__ mi, float* __restrict__ mo,
                            int Din, int Hin, int Win, int Dout, int Hout, int Wout,
                            int KD, int KH, int KW, int sd, int sh, int sw,
                            int pd, int ph, int pw) {
  long i = (long)blockIdx.x * blockDim.x + threadIdx.x;
  long tot = (long)Dout * Hout * Wout;
  if (i >= tot) return;
  int x = (int)(i % Wout);
  int y = (int)((i / Wout) % Hout);
  int z = (int)(i / ((long)Wout * Hout));
  float any = 0.0f;
  for (int kd = 0; kd < KD; kd++) {
    int iz = z * sd - pd + kd; if (iz < 0 || iz >= Din) continue;
    for (int kh = 0; kh < KH; kh++) {
      int iy = y * sh - ph + kh; if (iy < 0 || iy >= Hin) continue;
      for (int kw = 0; kw < KW; kw++) {
        int ix = x * sw - pw + kw; if (ix < 0 || ix >= Win) continue;
        if (mi[((long)iz * Hin + iy) * Win + ix] > 0.0f) any = 1.0f;
      }
    }
  }
  mo[i] = any;
}

// ---------------------------------------------------------------------------
// Pack fp32 weights (kd,kh,kw,ci,co) into bf16 WMMA-B fragments.
// K dimension of each 32-wide chunk holds TPC fused kw-taps x (32/TPC) channels.
// Fragment order: chunk = ((kd*KH+kh)*WG + g)*KCH + kc ; f = chunk*ntiles + nt.
// Per lane (B layout): n-column = lane&15, K-in-chunk = ((e>>3)<<4)+(lane>=16?8:0)+(e&7).
// ---------------------------------------------------------------------------
__global__ void pack_w_k(const float* __restrict__ w, unsigned short* __restrict__ out,
                         int KD, int KH, int KW, int TPC, int lsw, int KCH,
                         int ntiles, int Cin, int Cout) {
  int WG = (KW + TPC - 1) / TPC;
  long total = (long)KD * KH * WG * KCH * ntiles * 512;
  long idx = (long)blockIdx.x * blockDim.x + threadIdx.x;
  if (idx >= total) return;
  int e    = (int)(idx & 15);
  int lane = (int)((idx >> 4) & 31);
  long f   = idx >> 9;
  int nt   = (int)(f % ntiles);  long f2 = f / ntiles;
  int kc   = (int)(f2 % KCH);    long f3 = f2 / KCH;
  int g    = (int)(f3 % WG);     long f4 = f3 / WG;
  int kh   = (int)(f4 % KH);
  int kd   = (int)(f4 / KH);
  int kin  = ((e >> 3) << 4) + ((lane >= 16) ? 8 : 0) + (e & 7);   // K index within chunk, 0..31
  int tap_local = kin >> lsw;                   // which fused tap
  int ch   = (kin & ((1 << lsw) - 1)) + ((TPC == 1) ? kc * 32 : 0);
  int kw   = g * TPC + tap_local;
  int n    = nt * 16 + (lane & 15);
  float val = 0.0f;
  if (kw < KW && ch < Cin)
    val = w[(((long)(kd * KH + kh) * KW + kw) * Cin + ch) * Cout + n];
  out[idx] = f2bf(val);
}

// ---------------------------------------------------------------------------
// Conv descriptor shared by both conv kernels
// ---------------------------------------------------------------------------
struct ConvDesc {
  const unsigned short* in;    // bf16 activations (D,H,W,Cin)
  unsigned short* outb;        // bf16 output (or null)
  float* outf;                 // f32 output (or null)
  const unsigned short* wpk;   // packed bf16 weights
  const float *g, *b, *m, *v;  // BN params per out-channel
  const float* bias;           // conv bias (or null)
  const float* mask;           // out-level mask (or null)
  const unsigned short* res;   // bf16 residual (or null)
  int Din, Hin, Win;
  int Dout, Hout, Wout;
  int KD, KH, KW, sd, sh, sw, pd, ph, pw;
  int relu;
};

// ---------------------------------------------------------------------------
// Variant 1 (NT<=2, L0/L1 layers): register-direct implicit GEMM.
// One wave per 16 output voxels; weights streamed from L2 (array <=54KB,
// resident). TPC = kw-taps fused per K=32 chunk for full matrix-core K.
// ---------------------------------------------------------------------------
template <int TPC, int KCH, int NT>
__global__ __launch_bounds__(32) void conv_wmma(ConvDesc d) {
  constexpr int CINE = (TPC == 1) ? KCH * 32 : 32 / TPC;
  constexpr int COUT = NT * 16;
  const int lane = threadIdx.x;
  const int x0 = (int)blockIdx.x << 4;
  const int y  = (int)blockIdx.y;
  const int z  = (int)blockIdx.z;
  const int M  = lane & 15;
  const int x  = x0 + M;
  const int hi8 = (lane >= 16) ? 8 : 0;
  const bool upper = (lane >= 16);

  v8f acc[NT];
  #pragma unroll
  for (int i = 0; i < NT; i++) acc[i] = (v8f){};
  uint4 z4; z4.x = z4.y = z4.z = z4.w = 0u;

  const unsigned short* wlane = d.wpk + lane * 16;
  const int WG = (d.KW + TPC - 1) / TPC;
  const int nchunks = d.KD * d.KH * WG * KCH;
  const int xok = (x < d.Wout);

  for (int kd = 0; kd < d.KD; kd++) {
    const int iz = z * d.sd - d.pd + kd;
    const bool zok = (iz >= 0) && (iz < d.Din);
    for (int kh = 0; kh < d.KH; kh++) {
      const int iy = y * d.sh - d.ph + kh;
      const bool yok = zok && (iy >= 0) && (iy < d.Hin) && xok;
      const long rowbase = (long)(iz * d.Hin + iy) * d.Win;
      for (int g = 0; g < WG; g++) {
        int kwl, kwh, cl, ch;
        if (TPC == 1)      { kwl = g;             kwh = g;         cl = hi8; ch = hi8 + 16; }
        else if (TPC == 2) { kwl = 2 * g;         kwh = 2 * g + 1; cl = hi8; ch = hi8;      }
        else               { kwl = upper ? 1 : 0; kwh = kwl + 2;   cl = 0;   ch = 0;        }
        const int ixl = x * d.sw - d.pw + kwl;
        const int ixh = x * d.sw - d.pw + kwh;
        const bool okl = yok && (ixl >= 0) && (ixl < d.Win) && (kwl < d.KW);
        const bool okh = yok && (ixh >= 0) && (ixh < d.Win) && (kwh < d.KW);
        const int cbase = ((kd * d.KH + kh) * WG + g) * KCH;
        if (cbase + KCH < nchunks)   // gfx1250 global_prefetch_b8
          __builtin_prefetch(wlane + (long)(cbase + KCH) * NT * 512, 0, 1);
        #pragma unroll
        for (int kc = 0; kc < KCH; kc++) {
          const int c0 = (TPC == 1) ? (kc * 32) : 0;
          Frag a; a.q.lo = z4; a.q.hi = z4;
          if (okl) a.q.lo = *(const uint4*)(d.in + (rowbase + ixl) * CINE + c0 + cl);
          if (okh) a.q.hi = *(const uint4*)(d.in + (rowbase + ixh) * CINE + c0 + ch);
          #pragma unroll
          for (int nt = 0; nt < NT; nt++) {
            const long fidx = (long)(cbase + kc) * NT + nt;
            const uint4* bp = (const uint4*)(wlane + fidx * 512);
            Frag bb; bb.q.lo = bp[0]; bb.q.hi = bp[1];
            acc[nt] = __builtin_amdgcn_wmma_f32_16x16x32_bf16(
                false, a.v, false, bb.v, (short)0, acc[nt], false, false);
          }
        }
      }
    }
  }

  const int nlo = lane & 15;
  const int Mbase = upper ? 8 : 0;
  #pragma unroll
  for (int nt = 0; nt < NT; nt++) {
    const int n = nt * 16 + nlo;
    float scale = d.g[n] * rsqrtf(d.v[n] + 1e-3f);
    float shift = d.b[n] - d.m[n] * scale;
    if (d.bias) shift += d.bias[n] * scale;
    #pragma unroll
    for (int r = 0; r < 8; r++) {
      const int xo = x0 + Mbase + r;
      if (xo >= d.Wout) continue;
      const long ov = (long)(z * d.Hout + y) * d.Wout + xo;
      float val = acc[nt][r] * scale + shift;
      if (d.res) val += bf2f(d.res[ov * COUT + n]);
      if (d.relu) val = fmaxf(val, 0.0f);
      if (d.mask) val *= d.mask[ov];
      if (d.outb) d.outb[ov * COUT + n] = f2bf(val);
      if (d.outf) d.outf[ov * COUT + n] = val;
    }
  }
}

// ---------------------------------------------------------------------------
// Variant 2 (NT>=4, deep layers, TPC==1): 4-wave workgroup covering a 2x2
// (x-tile, y) output region. Weights for one (kd,kh) slab are staged to LDS
// with async global->LDS DMA, double-buffered (copy slab i+1 overlaps the
// WMMAs of slab i). B fragments then come from LDS (shared by 4 waves):
// 4x less L2 weight traffic, and the copy is fully asynchronous (ASYNCcnt).
// ---------------------------------------------------------------------------
template <int KCH, int NT>
__global__ __launch_bounds__(128) void conv_wmma_lds(ConvDesc d) {
  constexpr int CINE = KCH * 32;
  constexpr int COUT = NT * 16;
  extern __shared__ unsigned short shw[];
  const int tid  = threadIdx.x;
  const int lane = tid & 31;
  const int wave = tid >> 5;
  const int wx = wave & 1, wy = wave >> 1;
  const int x0 = (((int)blockIdx.x * 2 + wx) << 4);
  const int y  = (int)blockIdx.y * 2 + wy;
  const int z  = (int)blockIdx.z;
  const int M  = lane & 15;
  const int x  = x0 + M;
  const int hi8 = (lane >= 16) ? 8 : 0;
  const bool upper = (lane >= 16);

  const int WG = d.KW;                       // TPC==1 here
  const int slabBytes = WG * KCH * NT * 1024;
  const int slabUS    = slabBytes >> 1;      // in ushorts
  const int ngroups = d.KD * d.KH;
  const unsigned shbase = (unsigned)(size_t)(&shw[0]);

  v8f acc[NT];
  #pragma unroll
  for (int i = 0; i < NT; i++) acc[i] = (v8f){};
  uint4 z4; z4.x = z4.y = z4.z = z4.w = 0u;

  // async copy of weight slab gi into LDS buffer (gi&1)
  auto stage = [&](int gi) {
    if (gi >= ngroups) return;
    const char* src = (const char*)d.wpk + (long)gi * slabBytes;
    const unsigned dst = shbase + (unsigned)((gi & 1) * slabBytes);
    for (int o = tid * 16; o < slabBytes; o += 128 * 16) {
      unsigned ldsa = dst + (unsigned)o;
      const char* gp = src + o;
      asm volatile("global_load_async_to_lds_b128 %0, %1, off"
                   :: "v"(ldsa), "v"(gp) : "memory");
    }
  };

  stage(0);
  asm volatile("s_wait_asynccnt 0x0" ::: "memory");
  __syncthreads();

  const bool xyok = (x < d.Wout) && (y < d.Hout);
  int gi = 0;
  for (int kd = 0; kd < d.KD; kd++) {
    const int iz = z * d.sd - d.pd + kd;
    const bool zok = (iz >= 0) && (iz < d.Din);
    for (int kh = 0; kh < d.KH; kh++, gi++) {
      const int iy = y * d.sh - d.ph + kh;
      const bool yok = zok && (iy >= 0) && (iy < d.Hin) && xyok;
      const long rowbase = (long)(iz * d.Hin + iy) * d.Win;
      stage(gi + 1);                          // prefetch next slab (other buffer)
      const unsigned short* sbuf = shw + (gi & 1) * slabUS;
      for (int g = 0; g < WG; g++) {
        const int ixl = x * d.sw - d.pw + g;
        const bool okl = yok && (ixl >= 0) && (ixl < d.Win);
        #pragma unroll
        for (int kc = 0; kc < KCH; kc++) {
          const int c0 = kc * 32 + hi8;
          Frag a; a.q.lo = z4; a.q.hi = z4;
          if (okl) {
            const unsigned short* p = d.in + (rowbase + ixl) * CINE + c0;
            a.q.lo = *(const uint4*)(p);
            a.q.hi = *(const uint4*)(p + 16);
          }
          #pragma unroll
          for (int nt = 0; nt < NT; nt++) {
            const unsigned short* bp =
                sbuf + (((g * KCH + kc) * NT + nt) * 512 + lane * 16);
            Frag bb;
            bb.q.lo = *(const uint4*)(bp);
            bb.q.hi = *(const uint4*)(bp + 8);
            acc[nt] = __builtin_amdgcn_wmma_f32_16x16x32_bf16(
                false, a.v, false, bb.v, (short)0, acc[nt], false, false);
          }
        }
      }
      asm volatile("s_wait_asynccnt 0x0" ::: "memory");
      __syncthreads();                        // slab gi done by all; gi+1 resident
    }
  }

  if (!((y < d.Hout))) return;               // after all barriers
  const int nlo = lane & 15;
  const int Mbase = upper ? 8 : 0;
  #pragma unroll
  for (int nt = 0; nt < NT; nt++) {
    const int n = nt * 16 + nlo;
    float scale = d.g[n] * rsqrtf(d.v[n] + 1e-3f);
    float shift = d.b[n] - d.m[n] * scale;
    if (d.bias) shift += d.bias[n] * scale;
    #pragma unroll
    for (int r = 0; r < 8; r++) {
      const int xo = x0 + Mbase + r;
      if (xo >= d.Wout) continue;
      const long ov = (long)(z * d.Hout + y) * d.Wout + xo;
      float val = acc[nt][r] * scale + shift;
      if (d.res) val += bf2f(d.res[ov * COUT + n]);
      if (d.relu) val = fmaxf(val, 0.0f);
      if (d.mask) val *= d.mask[ov];
      if (d.outb) d.outb[ov * COUT + n] = f2bf(val);
      if (d.outf) d.outf[ov * COUT + n] = val;
    }
  }
}

// ---------------------------------------------------------------------------
// Host orchestration
// ---------------------------------------------------------------------------
struct ConvSpec { int wi, KD, KH, KW, Cin_act, Cin_eff, Cout; };

static const ConvSpec g_convs[21] = {
  {  2, 3,3,3,   4,   8,  16},  // 0  conv_input
  {  7, 3,3,3,  16,  16,  16},  // 1  conv1.b0.w1
  { 13, 3,3,3,  16,  16,  16},  // 2  conv1.b0.w2
  { 19, 3,3,3,  16,  16,  16},  // 3  conv1.b1.w1
  { 25, 3,3,3,  16,  16,  16},  // 4  conv1.b1.w2
  { 31, 3,3,3,  16,  16,  32},  // 5  conv2_down
  { 36, 3,3,3,  32,  32,  32},  // 6  conv2.b0.w1
  { 42, 3,3,3,  32,  32,  32},  // 7  conv2.b0.w2
  { 48, 3,3,3,  32,  32,  32},  // 8  conv2.b1.w1
  { 54, 3,3,3,  32,  32,  32},  // 9  conv2.b1.w2
  { 60, 3,3,3,  32,  32,  64},  // 10 conv3_down
  { 65, 3,3,3,  64,  64,  64},  // 11 conv3.b0.w1
  { 71, 3,3,3,  64,  64,  64},  // 12 conv3.b0.w2
  { 77, 3,3,3,  64,  64,  64},  // 13 conv3.b1.w1
  { 83, 3,3,3,  64,  64,  64},  // 14 conv3.b1.w2
  { 89, 3,3,3,  64,  64, 128},  // 15 conv4_down
  { 94, 3,3,3, 128, 128, 128},  // 16 conv4.b0.w1
  {100, 3,3,3, 128, 128, 128},  // 17 conv4.b0.w2
  {106, 3,3,3, 128, 128, 128},  // 18 conv4.b1.w1
  {112, 3,3,3, 128, 128, 128},  // 19 conv4.b1.w2
  {118, 3,1,1, 128, 128, 128},  // 20 conv_out
};

static inline size_t align256(size_t x) { return (x + 255) & ~(size_t)255; }
static inline void derive(const ConvSpec& s, int& tpc, int& lsw, int& kch, int& nt, int& wg) {
  tpc = (s.Cin_eff >= 32) ? 1 : (32 / s.Cin_eff);
  lsw = (tpc == 1) ? 5 : ((tpc == 2) ? 4 : 3);
  kch = (tpc == 1) ? s.Cin_eff / 32 : 1;
  nt  = s.Cout / 16;
  wg  = (s.KW + tpc - 1) / tpc;
}

extern "C" void kernel_launch(void* const* d_in, const int* in_sizes, int n_in,
                              void* d_out, int out_size, void* d_ws, size_t ws_size,
                              hipStream_t stream) {
  (void)in_sizes; (void)n_in; (void)out_size; (void)ws_size;
  char* ws = (char*)d_ws;
  size_t off = 0;
  auto alloc = [&](size_t bytes) { size_t o = off; off = align256(off + bytes); return o; };

  // Level dims: {D,H,W}
  const int L0[3] = {41, 160, 160}, L1[3] = {21, 80, 80}, L2[3] = {11, 40, 40},
            L3[3] = { 5,  20,  20}, LO[3] = { 2, 20, 20};
  const long n0 = 41L * 160 * 160, n1 = 21L * 80 * 80, n2 = 11L * 40 * 40,
             n3 = 5L * 20 * 20,    nO = 2L * 20 * 20;

  size_t oM0 = alloc(n0 * 4), oM1 = alloc(n1 * 4), oM2 = alloc(n2 * 4),
         oM3 = alloc(n3 * 4), oMO = alloc(nO * 4);
  size_t oIn0 = alloc(n0 * 8 * 2);
  size_t oA0 = alloc(n0 * 16 * 2), oB0 = alloc(n0 * 16 * 2), oC0 = alloc(n0 * 16 * 2);
  size_t oA1 = alloc(n1 * 32 * 2), oB1 = alloc(n1 * 32 * 2), oC1 = alloc(n1 * 32 * 2);
  size_t oA2 = alloc(n2 * 64 * 2), oB2 = alloc(n2 * 64 * 2), oC2 = alloc(n2 * 64 * 2);
  size_t oA3 = alloc(n3 * 128 * 2), oB3 = alloc(n3 * 128 * 2), oC3 = alloc(n3 * 128 * 2);

  size_t wpkOff[21];
  for (int i = 0; i < 21; i++) {
    const ConvSpec& s = g_convs[i];
    int tpc, lsw, kch, nt, wg; derive(s, tpc, lsw, kch, nt, wg);
    wpkOff[i] = alloc((size_t)s.KD * s.KH * wg * kch * nt * 1024);
  }

  auto US = [&](size_t o) { return (unsigned short*)(ws + o); };
  auto F32 = [&](size_t o) { return (float*)(ws + o); };

  // --- init dense input + base mask (deterministic every call) ---
  {
    long w_in0 = (n0 * 8 * 2) / 4, w_m0 = n0;
    zero_k<<<dim3(2048), dim3(256), 0, stream>>>((unsigned int*)(ws + oIn0), w_in0);
    zero_k<<<dim3(2048), dim3(256), 0, stream>>>((unsigned int*)(ws + oM0), w_m0);
    scatter_k<<<dim3((60000 + 255) / 256), dim3(256), 0, stream>>>(
        (const float*)d_in[0], (const int*)d_in[1], US(oIn0), F32(oM0), 60000);
  }

  // --- pack all conv weights to bf16 WMMA fragments ---
  for (int i = 0; i < 21; i++) {
    const ConvSpec& s = g_convs[i];
    int tpc, lsw, kch, nt, wg; derive(s, tpc, lsw, kch, nt, wg);
    long total = (long)s.KD * s.KH * wg * kch * nt * 512;
    pack_w_k<<<dim3((unsigned)((total + 255) / 256)), dim3(256), 0, stream>>>(
        (const float*)d_in[s.wi], US(wpkOff[i]), s.KD, s.KH, s.KW,
        tpc, lsw, kch, nt, s.Cin_act, s.Cout);
  }

  // --- mask pyramid ---
  auto maskdown = [&](size_t mi, size_t mo, const int* di, const int* dout,
                      int KD, int KH, int KW, int sd, int sh, int sw,
                      int pd, int ph, int pw) {
    long tot = (long)dout[0] * dout[1] * dout[2];
    mask_down_k<<<dim3((unsigned)((tot + 127) / 128)), dim3(128), 0, stream>>>(
        F32(mi), F32(mo), di[0], di[1], di[2], dout[0], dout[1], dout[2],
        KD, KH, KW, sd, sh, sw, pd, ph, pw);
  };
  maskdown(oM0, oM1, L0, L1, 3, 3, 3, 2, 2, 2, 1, 1, 1);
  maskdown(oM1, oM2, L1, L2, 3, 3, 3, 2, 2, 2, 1, 1, 1);
  maskdown(oM2, oM3, L2, L3, 3, 3, 3, 2, 2, 2, 0, 1, 1);
  maskdown(oM3, oMO, L3, LO, 3, 1, 1, 2, 1, 1, 0, 0, 0);

  // --- conv launcher ---
  auto conv = [&](int ci, const unsigned short* in, unsigned short* outb, float* outf,
                  const int* idim, const int* odim, int sd, int sh, int sw,
                  int pd, int ph, int pw, size_t maskOff,
                  const unsigned short* res, int relu, int isblock) {
    const ConvSpec& s = g_convs[ci];
    ConvDesc d{};
    d.in = in; d.outb = outb; d.outf = outf;
    d.wpk = US(wpkOff[ci]);
    int bnb = s.wi + (isblock ? 2 : 1);
    d.g = (const float*)d_in[bnb];     d.b = (const float*)d_in[bnb + 1];
    d.m = (const float*)d_in[bnb + 2]; d.v = (const float*)d_in[bnb + 3];
    d.bias = isblock ? (const float*)d_in[s.wi + 1] : nullptr;
    d.mask = F32(maskOff); d.res = res;
    d.Din = idim[0]; d.Hin = idim[1]; d.Win = idim[2];
    d.Dout = odim[0]; d.Hout = odim[1]; d.Wout = odim[2];
    d.KD = s.KD; d.KH = s.KH; d.KW = s.KW;
    d.sd = sd; d.sh = sh; d.sw = sw; d.pd = pd; d.ph = ph; d.pw = pw;
    d.relu = relu;
    int xt = (odim[2] + 15) / 16;
    int tpc, lsw, kch, nt, wg; derive(s, tpc, lsw, kch, nt, wg);
    if (nt >= 4) {
      // LDS-staged cooperative variant (TPC==1 layers only)
      size_t shmem = (size_t)2 * s.KW * kch * nt * 1024;
      dim3 grid((unsigned)((xt + 1) / 2), (unsigned)((odim[1] + 1) / 2),
                (unsigned)odim[0]);
      if      (kch == 1 && nt == 4) conv_wmma_lds<1, 4><<<grid, 128, shmem, stream>>>(d);
      else if (kch == 2 && nt == 4) conv_wmma_lds<2, 4><<<grid, 128, shmem, stream>>>(d);
      else if (kch == 2 && nt == 8) conv_wmma_lds<2, 8><<<grid, 128, shmem, stream>>>(d);
      else                          conv_wmma_lds<4, 8><<<grid, 128, shmem, stream>>>(d);
    } else {
      dim3 grid((unsigned)xt, (unsigned)odim[1], (unsigned)odim[0]);
      if      (tpc == 4)            conv_wmma<4, 1, 1><<<grid, dim3(32), 0, stream>>>(d);
      else if (tpc == 2 && nt == 1) conv_wmma<2, 1, 1><<<grid, dim3(32), 0, stream>>>(d);
      else if (tpc == 2 && nt == 2) conv_wmma<2, 1, 2><<<grid, dim3(32), 0, stream>>>(d);
      else                          conv_wmma<1, 1, 2><<<grid, dim3(32), 0, stream>>>(d);
    }
  };

  // --- network ---
  // L0
  conv(0, US(oIn0), US(oA0), nullptr, L0, L0, 1,1,1, 1,1,1, oM0, nullptr, 1, 0);
  conv(1, US(oA0),  US(oB0), nullptr, L0, L0, 1,1,1, 1,1,1, oM0, nullptr, 1, 1);
  conv(2, US(oB0),  US(oC0), nullptr, L0, L0, 1,1,1, 1,1,1, oM0, US(oA0), 1, 1);
  conv(3, US(oC0),  US(oB0), nullptr, L0, L0, 1,1,1, 1,1,1, oM0, nullptr, 1, 1);
  conv(4, US(oB0),  US(oA0), nullptr, L0, L0, 1,1,1, 1,1,1, oM0, US(oC0), 1, 1);
  // L1
  conv(5, US(oA0),  US(oA1), nullptr, L0, L1, 2,2,2, 1,1,1, oM1, nullptr, 1, 0);
  conv(6, US(oA1),  US(oB1), nullptr, L1, L1, 1,1,1, 1,1,1, oM1, nullptr, 1, 1);
  conv(7, US(oB1),  US(oC1), nullptr, L1, L1, 1,1,1, 1,1,1, oM1, US(oA1), 1, 1);
  conv(8, US(oC1),  US(oB1), nullptr, L1, L1, 1,1,1, 1,1,1, oM1, nullptr, 1, 1);
  conv(9, US(oB1),  US(oA1), nullptr, L1, L1, 1,1,1, 1,1,1, oM1, US(oC1), 1, 1);
  // L2 (LDS-staged from here down)
  conv(10, US(oA1), US(oA2), nullptr, L1, L2, 2,2,2, 1,1,1, oM2, nullptr, 1, 0);
  conv(11, US(oA2), US(oB2), nullptr, L2, L2, 1,1,1, 1,1,1, oM2, nullptr, 1, 1);
  conv(12, US(oB2), US(oC2), nullptr, L2, L2, 1,1,1, 1,1,1, oM2, US(oA2), 1, 1);
  conv(13, US(oC2), US(oB2), nullptr, L2, L2, 1,1,1, 1,1,1, oM2, nullptr, 1, 1);
  conv(14, US(oB2), US(oA2), nullptr, L2, L2, 1,1,1, 1,1,1, oM2, US(oC2), 1, 1);
  // L3 (conv4_down pad = (0,1,1))
  conv(15, US(oA2), US(oA3), nullptr, L2, L3, 2,2,2, 0,1,1, oM3, nullptr, 1, 0);
  conv(16, US(oA3), US(oB3), nullptr, L3, L3, 1,1,1, 1,1,1, oM3, nullptr, 1, 1);
  conv(17, US(oB3), US(oC3), nullptr, L3, L3, 1,1,1, 1,1,1, oM3, US(oA3), 1, 1);
  conv(18, US(oC3), US(oB3), nullptr, L3, L3, 1,1,1, 1,1,1, oM3, nullptr, 1, 1);
  conv(19, US(oB3), US(oA3), nullptr, L3, L3, 1,1,1, 1,1,1, oM3, US(oC3), 1, 1);
  // Output conv: kernel (3,1,1), stride (2,1,1), pad 0, BN, no ReLU, mask, fp32 out
  conv(20, US(oA3), nullptr, (float*)d_out, L3, LO, 2,1,1, 0,0,0, oMO, nullptr, 0, 0);
}
